// Attention_37907381355196
// MI455X (gfx1250) — compile-verified, matrix operand order checked by reading, not run on hardware
//
#include <hip/hip_runtime.h>
#include <hip/hip_bf16.h>
#include <stdint.h>

// ---------------------------------------------------------------------------
// Attention block for MI455X (gfx1250, wave32, WMMA).
//   fp32->bf16 convert (+weight transpose)
//   -> QKV GEMM (async global->LDS double-buffered, WMMA bf16) w/ fused RoPE
//   -> flash attention (online softmax, WMMA QK^T and PV)
//   -> output GEMM (fp32 store).
// ---------------------------------------------------------------------------

typedef __bf16 bf16;
typedef __bf16 v16bf  __attribute__((ext_vector_type(16)));
typedef __bf16 bf16x8 __attribute__((ext_vector_type(8)));
typedef float  v8f    __attribute__((ext_vector_type(8)));
typedef int    v4i    __attribute__((ext_vector_type(4)));

#define AS1 __attribute__((address_space(1)))
#define AS3 __attribute__((address_space(3)))

#define B_   4
#define S_   2048
#define D_   768
#define H_   12
#define KVH_ 4
#define HD_  64
#define M_   (B_ * S_)       // 8192 rows
#define TK_  32              // K step per WMMA round
#define LDA_ 40              // padded LDS row (elements): 80B rows -> conflict-free b128 reads

// ---------------------------------------------------------------------------
// 16-bit A/B fragment load for V_WMMA_*_16X16X32_BF16 (ISA 7.12.2):
// lanes 0-15 hold row m=lane at K {0..7, 16..23}; lanes 16-31 same rows at
// K {8..15, 24..31}.  Row-major [rows][K] source => two 16B loads per lane.
// ---------------------------------------------------------------------------
__device__ __forceinline__ v16bf load_frag(const bf16* base, int ld, int lane) {
  const int m    = lane & 15;
  const int half = lane >> 4;
  const bf16* p  = base + (size_t)m * ld + half * 8;
  bf16x8 lo = *(const bf16x8*)(p);
  bf16x8 hi = *(const bf16x8*)(p + 16);
  v16bf f;
#pragma unroll
  for (int j = 0; j < 8; ++j) { f[j] = lo[j]; f[j + 8] = hi[j]; }
  return f;
}

__device__ __forceinline__ v8f wmma_bf16(v16bf a, v16bf b, v8f c) {
  return __builtin_amdgcn_wmma_f32_16x16x32_bf16(false, a, false, b,
                                                 (short)0, c, false, false);
}

// ---------------------------------------------------------------------------
// CDNA5 async global->LDS copy (ASYNCcnt-tracked), with safe fallback.
// Builtin signature (from toolchain diagnostic): (AS1 v4i* src, AS3 v4i* dst,
// imm offset, imm cpol).
// ---------------------------------------------------------------------------
__device__ __forceinline__ void async_b128(const bf16* g, bf16* l) {
#if __has_builtin(__builtin_amdgcn_global_load_async_to_lds_b128)
  __builtin_amdgcn_global_load_async_to_lds_b128(
      (AS1 v4i*)g, (AS3 v4i*)l, 0, 0);
#else
  *(bf16x8*)l = *(const bf16x8*)g;   // sync fallback: load + ds_store
#endif
}

template <int N>
__device__ __forceinline__ void wait_async() {
#if __has_builtin(__builtin_amdgcn_global_load_async_to_lds_b128)
#if __has_builtin(__builtin_amdgcn_s_wait_asynccnt)
  __builtin_amdgcn_s_wait_asynccnt(N);
#else
  if (N == 0) { __asm volatile("s_wait_asynccnt 0x0" ::: "memory"); }
  else        { __asm volatile("s_wait_asynccnt 0x4" ::: "memory"); }
#endif
#endif
}

// ---------------------------------------------------------------------------
// Converters
// ---------------------------------------------------------------------------
__global__ void cvt_kernel(const float* __restrict__ src, bf16* __restrict__ dst, int n) {
  int i = blockIdx.x * blockDim.x + threadIdx.x;
  if (i < n) dst[i] = (bf16)src[i];
}

// dst[c][r] = src[r][c]  (weights are [K][N] in the reference; we want [N][K])
__global__ void cvtT_kernel(const float* __restrict__ src, bf16* __restrict__ dst,
                            int R, int C) {
  int i = blockIdx.x * blockDim.x + threadIdx.x;
  if (i < R * C) {
    int r = i / C, c = i % C;
    dst[(size_t)c * R + r] = (bf16)src[i];
  }
}

// ---------------------------------------------------------------------------
// Projection GEMM: C[M][N] = A[M][K] * WT[N][K]^T   (bf16 in, f32 acc)
// Block = 256 threads (8 waves), tile 128M x 128N, waves 4x2, wave = 32x64.
// A/B tiles staged in LDS via async copies, double-buffered.
// Epilogue modes:
//   0: Q -> RoPE, bf16 [B][H][S][HD]      1: K -> RoPE, bf16 [B][KVH][S][HD]
//   2: V -> bf16 transposed [B][KVH][HD][S]  3: O -> fp32 [M][N]
// ---------------------------------------------------------------------------
__global__ __launch_bounds__(256)
void proj_kernel(const bf16* __restrict__ A, const bf16* __restrict__ WT,
                 const float* __restrict__ fc, const float* __restrict__ fs,
                 void* __restrict__ out, int N, int K, int mode) {
  __shared__ bf16 sA[2][128 * LDA_];
  __shared__ bf16 sB[2][128 * LDA_];

  const int tid  = threadIdx.x;
  const int lane = tid & 31;
  const int wave = tid >> 5;
  const int wm   = wave >> 1;                 // 0..3
  const int wn   = wave & 1;                  // 0..1
  const int rb   = blockIdx.x * 128;          // block row base (M)
  const int nb   = blockIdx.y * 128;          // block col base (N)
  const int r0   = wm * 32;                   // wave rows within tile
  const int c0   = wn * 64;                   // wave cols within tile
  const int ksteps = K / TK_;

  // Tile copy: 512 16B chunks per operand tile, 2 per thread per operand.
  auto issue = [&](int buf, int kstep) {
    const int k0 = kstep * TK_;
#pragma unroll
    for (int i = 0; i < 2; ++i) {
      const int c   = tid + i * 256;
      const int row = c >> 2;
      const int kc  = (c & 3) * 8;
      async_b128(A  + (size_t)(rb + row) * K + k0 + kc, &sA[buf][row * LDA_ + kc]);
      async_b128(WT + (size_t)(nb + row) * K + k0 + kc, &sB[buf][row * LDA_ + kc]);
    }
  };

  v8f acc[2][4];
#pragma unroll
  for (int i = 0; i < 2; ++i)
#pragma unroll
    for (int j = 0; j < 4; ++j)
#pragma unroll
      for (int r = 0; r < 8; ++r) acc[i][j][r] = 0.0f;

  issue(0, 0);
  for (int ks = 0; ks < ksteps; ++ks) {
    const int buf = ks & 1;
    if (ks + 1 < ksteps) { issue(buf ^ 1, ks + 1); wait_async<4>(); }
    else                 { wait_async<0>(); }
    __syncthreads();

    v16bf a0 = load_frag(&sA[buf][(r0 +  0) * LDA_], LDA_, lane);
    v16bf a1 = load_frag(&sA[buf][(r0 + 16) * LDA_], LDA_, lane);
#pragma unroll
    for (int j = 0; j < 4; ++j) {
      v16bf bfr = load_frag(&sB[buf][(c0 + j * 16) * LDA_], LDA_, lane);
      acc[0][j] = wmma_bf16(a0, bfr, acc[0][j]);
      acc[1][j] = wmma_bf16(a1, bfr, acc[1][j]);
    }
    __syncthreads();   // all waves done reading buf before it is re-filled
  }

  // Epilogue.  C layout: lane n = lane&15, rows m = r + 8*(lane>>4).
  const int n    = lane & 15;
  const int half = lane >> 4;
#pragma unroll
  for (int i = 0; i < 2; ++i) {
#pragma unroll
    for (int j = 0; j < 4; ++j) {
#pragma unroll
      for (int r = 0; r < 8; ++r) {
        const int row = rb + r0 + i * 16 + r + 8 * half;   // = b*S + s
        const int col = nb + c0 + j * 16 + n;              // = h*HD + d
        float v = acc[i][j][r];
        if (mode <= 1) {
          // RoPE on f32 accumulator: even/odd head-dims sit in adjacent lanes.
          const int s = row & (S_ - 1);
          const int d = col & (HD_ - 1);
          const float other = __shfl_xor(v, 1, 32);
          const float cc = fc[s * (HD_ / 2) + (d >> 1)];
          const float sn = fs[s * (HD_ / 2) + (d >> 1)];
          v = ((d & 1) == 0) ? (v * cc - other * sn) : (other * sn + v * cc);
        }
        if (mode == 0) {
          const int b = row / S_, s = row % S_;
          const int h = col / HD_, d = col % HD_;
          ((bf16*)out)[(((size_t)b * H_ + h) * S_ + s) * HD_ + d] = (bf16)v;
        } else if (mode == 1) {
          const int b = row / S_, s = row % S_;
          const int h = col / HD_, d = col % HD_;
          ((bf16*)out)[(((size_t)b * KVH_ + h) * S_ + s) * HD_ + d] = (bf16)v;
        } else if (mode == 2) {
          const int b = row / S_, s = row % S_;
          const int h = col / HD_, d = col % HD_;
          ((bf16*)out)[(((size_t)b * KVH_ + h) * HD_ + d) * S_ + s] = (bf16)v;
        } else {
          ((float*)out)[(size_t)row * N + col] = v;
        }
      }
    }
  }
}

// ---------------------------------------------------------------------------
// Flash attention: block = 4 waves = 64 q rows of one (b, h); wave = 16 rows.
// Key tiles of 32; online softmax; P staged through a per-wave LDS strip to
// convert C-layout f32 -> A-layout bf16 for the PV WMMA.
// ---------------------------------------------------------------------------
__global__ __launch_bounds__(128)
void flash_kernel(const bf16* __restrict__ Q, const bf16* __restrict__ Kmat,
                  const bf16* __restrict__ VT, const int* __restrict__ mask,
                  bf16* __restrict__ O) {
  __shared__ bf16 ldsP[4][16][40];

  const int lane = threadIdx.x & 31;
  const int wave = threadIdx.x >> 5;
  int bid = blockIdx.x;
  const int qt  = bid % (S_ / 64); bid /= (S_ / 64);
  const int h   = bid % H_;
  const int b   = bid / H_;
  const int kvh = h / (H_ / KVH_);
  const int q0  = qt * 64 + wave * 16;

  const bf16* qbase = Q  + (((size_t)b * H_   + h)   * S_ + q0) * HD_;
  const bf16* kb    = Kmat + ((size_t)b * KVH_ + kvh) * S_ * HD_;
  const bf16* vtb   = VT + ((size_t)b * KVH_ + kvh) * (size_t)HD_ * S_;

  const v16bf qf0 = load_frag(qbase,      HD_, lane);
  const v16bf qf1 = load_frag(qbase + 32, HD_, lane);

  v8f   o[4];
  float mi[8], li[8];
#pragma unroll
  for (int j = 0; j < 4; ++j)
#pragma unroll
    for (int r = 0; r < 8; ++r) o[j][r] = 0.0f;
#pragma unroll
  for (int r = 0; r < 8; ++r) { mi[r] = -__builtin_inff(); li[r] = 0.0f; }

  const int   n      = lane & 15;
  const int   half   = lane >> 4;
  const float scale  = 0.125f;               // 1/sqrt(HD)
  const int   ntiles = (q0 + 16 + 31) / 32;  // causal: keys <= q0+15

  for (int kt = 0; kt < ntiles; ++kt) {
    const int k0 = kt * 32;

    v8f s0, s1;
#pragma unroll
    for (int r = 0; r < 8; ++r) { s0[r] = 0.0f; s1[r] = 0.0f; }
    v16bf kf;
    kf = load_frag(kb + (size_t)k0 * HD_,             HD_, lane);
    s0 = wmma_bf16(qf0, kf, s0);
    kf = load_frag(kb + (size_t)k0 * HD_ + 32,        HD_, lane);
    s0 = wmma_bf16(qf1, kf, s0);
    kf = load_frag(kb + (size_t)(k0 + 16) * HD_,      HD_, lane);
    s1 = wmma_bf16(qf0, kf, s1);
    kf = load_frag(kb + (size_t)(k0 + 16) * HD_ + 32, HD_, lane);
    s1 = wmma_bf16(qf1, kf, s1);

    const int  key0 = k0 + n;
    const int  key1 = k0 + 16 + n;
    const bool mv0  = mask[b * S_ + key0] != 0;
    const bool mv1  = mask[b * S_ + key1] != 0;

    float alpha[8];
#pragma unroll
    for (int r = 0; r < 8; ++r) {
      const int qi = q0 + r + 8 * half;
      float v0 = (mv0 && key0 <= qi) ? s0[r] * scale : -__builtin_inff();
      float v1 = (mv1 && key1 <= qi) ? s1[r] * scale : -__builtin_inff();
      float rm = fmaxf(v0, v1);
#pragma unroll
      for (int w = 1; w < 16; w <<= 1) rm = fmaxf(rm, __shfl_xor(rm, w, 32));
      const float mn = fmaxf(mi[r], rm);
      const float al = (mn == -__builtin_inff()) ? 1.0f : __expf(mi[r] - mn);
      const float p0 = (v0 == -__builtin_inff()) ? 0.0f : __expf(v0 - mn);
      const float p1 = (v1 == -__builtin_inff()) ? 0.0f : __expf(v1 - mn);
      float ps = p0 + p1;
#pragma unroll
      for (int w = 1; w < 16; w <<= 1) ps += __shfl_xor(ps, w, 32);
      li[r]    = li[r] * al + ps;
      mi[r]    = mn;
      alpha[r] = al;
      const int m = r + 8 * half;
      ldsP[wave][m][n]      = (bf16)p0;
      ldsP[wave][m][n + 16] = (bf16)p1;
    }
#pragma unroll
    for (int j = 0; j < 4; ++j)
#pragma unroll
      for (int r = 0; r < 8; ++r) o[j][r] *= alpha[r];

    // Per-wave LDS RAW: wait for our ds_stores before reloading as A fragment.
    __asm volatile("s_wait_dscnt 0x0" ::: "memory");

    const v16bf pf = load_frag(&ldsP[wave][0][0], 40, lane);
#pragma unroll
    for (int j = 0; j < 4; ++j) {
      v16bf vf = load_frag(vtb + (size_t)(j * 16) * S_ + k0, S_, lane);
      o[j] = wmma_bf16(pf, vf, o[j]);
    }
  }

#pragma unroll
  for (int j = 0; j < 4; ++j) {
#pragma unroll
    for (int r = 0; r < 8; ++r) {
      const int   qi  = q0 + r + 8 * half;
      const float inv = (li[r] > 0.0f) ? 1.0f / li[r] : 0.0f;
      O[((size_t)b * S_ + qi) * (H_ * HD_) + h * HD_ + j * 16 + n] =
          (bf16)(o[j][r] * inv);
    }
  }
}

// ---------------------------------------------------------------------------
// Host-side launch
// ---------------------------------------------------------------------------
extern "C" void kernel_launch(void* const* d_in, const int* in_sizes, int n_in,
                              void* d_out, int out_size, void* d_ws, size_t ws_size,
                              hipStream_t stream) {
  const float* x    = (const float*)d_in[0];
  const float* wq   = (const float*)d_in[1];
  const float* wk   = (const float*)d_in[2];
  const float* wv   = (const float*)d_in[3];
  const float* wo   = (const float*)d_in[4];
  const float* fc   = (const float*)d_in[5];
  const float* fs   = (const float*)d_in[6];
  const int*   mask = (const int*)d_in[7];
  (void)in_sizes; (void)n_in; (void)out_size; (void)ws_size;

  char* p = (char*)d_ws;
  auto alloc = [&](size_t bytes) -> char* {
    char* r = p;
    p += (bytes + 255) & ~(size_t)255;
    return r;
  };
  bf16* xbf  = (bf16*)alloc((size_t)M_ * D_ * 2);
  bf16* wqT  = (bf16*)alloc((size_t)D_ * D_ * 2);
  bf16* wkT  = (bf16*)alloc((size_t)(KVH_ * HD_) * D_ * 2);
  bf16* wvT  = (bf16*)alloc((size_t)(KVH_ * HD_) * D_ * 2);
  bf16* woT  = (bf16*)alloc((size_t)D_ * D_ * 2);
  bf16* qb   = (bf16*)alloc((size_t)M_ * (H_ * HD_) * 2);
  bf16* kbuf = (bf16*)alloc((size_t)B_ * KVH_ * S_ * HD_ * 2);
  bf16* vT   = (bf16*)alloc((size_t)B_ * KVH_ * HD_ * S_ * 2);
  bf16* attn = (bf16*)alloc((size_t)M_ * (H_ * HD_) * 2);

  int n;
  n = M_ * D_;
  cvt_kernel<<<(n + 255) / 256, 256, 0, stream>>>(x, xbf, n);
  n = D_ * D_;
  cvtT_kernel<<<(n + 255) / 256, 256, 0, stream>>>(wq, wqT, D_, D_);
  n = D_ * (KVH_ * HD_);
  cvtT_kernel<<<(n + 255) / 256, 256, 0, stream>>>(wk, wkT, D_, KVH_ * HD_);
  cvtT_kernel<<<(n + 255) / 256, 256, 0, stream>>>(wv, wvT, D_, KVH_ * HD_);
  n = D_ * D_;
  cvtT_kernel<<<(n + 255) / 256, 256, 0, stream>>>(wo, woT, D_, D_);

  const dim3 blk(256);
  proj_kernel<<<dim3(M_ / 128, (H_ * HD_) / 128),   blk, 0, stream>>>(
      xbf, wqT, fc, fs, qb,   H_ * HD_,   D_, 0);
  proj_kernel<<<dim3(M_ / 128, (KVH_ * HD_) / 128), blk, 0, stream>>>(
      xbf, wkT, fc, fs, kbuf, KVH_ * HD_, D_, 1);
  proj_kernel<<<dim3(M_ / 128, (KVH_ * HD_) / 128), blk, 0, stream>>>(
      xbf, wvT, fc, fs, vT,   KVH_ * HD_, D_, 2);

  flash_kernel<<<dim3((S_ / 64) * B_ * H_), dim3(128), 0, stream>>>(
      qb, kbuf, vT, mask, attn);

  proj_kernel<<<dim3(M_ / 128, D_ / 128), blk, 0, stream>>>(
      attn, woT, fc, fs, d_out, D_, H_ * HD_, 3);
}
